// ResidualBilinearDownProjRMSNorm_87531433492483
// MI455X (gfx1250) — matrix operand order repaired and checked by reading, not compiled, not run on hardware
//
#include <hip/hip_runtime.h>

// ---------------------------------------------------------------------------
// ResidualBilinearDownProjRMSNorm for MI455X (gfx1250)
//
// Roofline: 51.5 GFLOP embed GEMM vs 805 MB of x => 64 FLOP/B, HBM floor
// ~35us @ 23.3 TB/s => need >1.5 PFLOPS => bf16 WMMA (v_wmma_f32_16x16x32_bf16)
// with fp32 accumulation. All weights packed once to bf16 B-fragment layout.
// ---------------------------------------------------------------------------

typedef __attribute__((ext_vector_type(16))) __bf16 v16bf;
typedef __attribute__((ext_vector_type(8)))  float  v8f;

#define KIN   3072
#define DRES  128
#define RANK  64
#define NOUT  10
#define MT    64          // rows per workgroup
#define EPSV  1e-6f

__device__ __forceinline__ v8f wmma_bf16(v16bf a, v16bf b, v8f c) {
  // 8 args: (neg_a, A, neg_b, B, c_mod, C, reuse_a, reuse_b)
  return __builtin_amdgcn_wmma_f32_16x16x32_bf16(false, a, false, b, (short)0, c,
                                                 false, false);
}

// ---------------------------------------------------------------------------
// Pack torch-style W [Ntot x Ktot] (out, in) into bf16 WMMA B-fragments for
// B = W^T (K x N).  Fragment layout (ISA 7.12.2, 16-bit B 32x16):
//   dst[ ((kc*ntiles + nt)*32 + lane)*16 + i ]  holds
//   K = kc*32 + ((lane&16)>>1) + (i&7) + ((i&8)<<1) ,  N = nt*16 + (lane&15)
// so each lane's 16 bf16 load with two b128s in the GEMM inner loop.
// ---------------------------------------------------------------------------
__global__ void pack_b_frags(const float* __restrict__ W, __bf16* __restrict__ dst,
                             int Ktot, int Ntot) {
  int idx = blockIdx.x * blockDim.x + threadIdx.x;
  int total = Ktot * Ntot;
  if (idx >= total) return;
  int i      = idx & 15;
  int lane   = (idx >> 4) & 31;
  int rest   = idx >> 9;
  int ntiles = Ntot >> 4;
  int nt = rest % ntiles;
  int kc = rest / ntiles;
  int K = kc * 32 + ((lane & 16) >> 1) + (i & 7) + ((i & 8) << 1);
  int N = nt * 16 + (lane & 15);
  dst[idx] = (__bf16)W[(size_t)N * Ktot + K];
}

// ---------------------------------------------------------------------------
// Fused:  r0 = x @ We^T ; two residual bilinear blocks ; out = r2 @ Wu^T
// 256 threads = 8 waves. Waves form a 4(M) x 2(N) grid over a 64x128 C tile;
// each wave owns 4 v8f accumulator tiles (16x64 strip).
// ---------------------------------------------------------------------------
__global__ __launch_bounds__(256) void fused_mlp(
    const float* __restrict__ x,
    const __bf16* __restrict__ Wb,
    const float* __restrict__ g1, const __bf16* __restrict__ L1b,
    const __bf16* __restrict__ R1b, const __bf16* __restrict__ D1b,
    const float* __restrict__ g2, const __bf16* __restrict__ L2b,
    const __bf16* __restrict__ R2b, const __bf16* __restrict__ D2b,
    const float* __restrict__ Wu,
    float* __restrict__ out)
{
  __shared__ float lds_r [MT][DRES + 4];   // residual stream (fp32)
  __shared__ float lds_xn[MT][DRES + 4];   // rmsnorm(r)*g
  __shared__ float lds_u [MT][RANK + 4];   // x @ L^T
  __shared__ float lds_v [MT][RANK + 4];   // x @ R^T

  const int tid   = threadIdx.x;
  const int wave  = tid >> 5;
  const int lane  = tid & 31;
  const int mwave = wave & 3;      // M group (16 rows)
  const int nwave = wave >> 2;     // N group (64 cols) for 128-wide outputs
  const int lhalf = lane >> 4;     // 0/1 lane-half
  const int l15   = lane & 15;
  const int koff  = lhalf ? 8 : 0; // A-fragment K base per lane half

  const int rowBase = blockIdx.x * MT;

  v8f acc[4];
  #pragma unroll
  for (int j = 0; j < 4; ++j) acc[j] = v8f{0.f, 0.f, 0.f, 0.f, 0.f, 0.f, 0.f, 0.f};

  // ------------------------- embed GEMM: r0 = x @ We^T --------------------
  const float* aptr = x + (size_t)(rowBase + mwave * 16 + l15) * KIN;
  for (int kc = 0; kc < KIN / 32; ++kc) {
    // A fragment: lane half 0 -> K {0..7,16..23}; half 1 -> {8..15,24..31}
    const float* a0p = aptr + kc * 32 + koff;
    float4 a01 = *(const float4*)(a0p);
    float4 a23 = *(const float4*)(a0p + 4);
    float4 a45 = *(const float4*)(a0p + 16);
    float4 a67 = *(const float4*)(a0p + 20);
    v16bf af;
    af[0]=(__bf16)a01.x; af[1]=(__bf16)a01.y; af[2]=(__bf16)a01.z; af[3]=(__bf16)a01.w;
    af[4]=(__bf16)a23.x; af[5]=(__bf16)a23.y; af[6]=(__bf16)a23.z; af[7]=(__bf16)a23.w;
    af[8]=(__bf16)a45.x; af[9]=(__bf16)a45.y; af[10]=(__bf16)a45.z; af[11]=(__bf16)a45.w;
    af[12]=(__bf16)a67.x; af[13]=(__bf16)a67.y; af[14]=(__bf16)a67.z; af[15]=(__bf16)a67.w;

    const __bf16* bp = Wb + (((size_t)kc * 8 + nwave * 4) * 32 + lane) * 16;
    #pragma unroll
    for (int j = 0; j < 4; ++j) {
      v16bf bf = *(const v16bf*)(bp + (size_t)j * 512);
      acc[j] = wmma_bf16(af, bf, acc[j]);
    }
  }

  // C layout (32-bit 16x16): lane l, vgpr e -> M = e + 8*(l>>4), N = l&15
  const int crow  = mwave * 16 + lhalf * 8;
  const int ccolb = nwave * 64;
  #pragma unroll
  for (int j = 0; j < 4; ++j)
    #pragma unroll
    for (int e = 0; e < 8; ++e)
      lds_r[crow + e][ccolb + j * 16 + l15] = acc[j][e];
  __syncthreads();

  const float*  gs[2] = {g1, g2};
  const __bf16* Ls[2] = {L1b, L2b};
  const __bf16* Rs[2] = {R1b, R2b};
  const __bf16* Dsb[2] = {D1b, D2b};

  for (int blk = 0; blk < 2; ++blk) {
    // ---- RMSNorm: 4 threads per row, shfl_xor reduce, then xn = r*rinv*g --
    {
      const int row = tid >> 2;
      const int seg = tid & 3;
      const float* rp = &lds_r[row][seg * 32];
      float ss = 0.f;
      #pragma unroll
      for (int k = 0; k < 32; ++k) { float t = rp[k]; ss += t * t; }
      ss += __shfl_xor(ss, 1);
      ss += __shfl_xor(ss, 2);
      float rinv = rsqrtf(ss * (1.f / (float)DRES) + EPSV);
      const float* g = gs[blk];
      #pragma unroll
      for (int k = 0; k < 32; ++k)
        lds_xn[row][seg * 32 + k] = rp[k] * rinv * g[seg * 32 + k];
    }
    __syncthreads();

    // ---- u = xn @ L^T (waves 0-3), v = xn @ R^T (waves 4-7): [64x64], K=128
    {
      const __bf16* Pb = (wave < 4) ? Ls[blk] : Rs[blk];
      const int m = mwave * 16 + l15;
      v8f uacc[4];
      #pragma unroll
      for (int j = 0; j < 4; ++j) uacc[j] = v8f{0.f,0.f,0.f,0.f,0.f,0.f,0.f,0.f};
      for (int kc = 0; kc < 4; ++kc) {
        const int kb = kc * 32 + koff;
        v16bf af;
        #pragma unroll
        for (int i = 0; i < 8; ++i) af[i]     = (__bf16)lds_xn[m][kb + i];
        #pragma unroll
        for (int i = 0; i < 8; ++i) af[8 + i] = (__bf16)lds_xn[m][kb + 16 + i];
        const __bf16* bp = Pb + (((size_t)kc * 4) * 32 + lane) * 16;
        #pragma unroll
        for (int j = 0; j < 4; ++j)
          uacc[j] = wmma_bf16(af, *(const v16bf*)(bp + (size_t)j * 512), uacc[j]);
      }
      float (*dstp)[RANK + 4] = (wave < 4) ? lds_u : lds_v;
      #pragma unroll
      for (int j = 0; j < 4; ++j)
        #pragma unroll
        for (int e = 0; e < 8; ++e)
          dstp[mwave * 16 + lhalf * 8 + e][j * 16 + l15] = uacc[j][e];
    }
    __syncthreads();

    // ---- r += (u .* v) @ D^T : K=64, out [64x128], accumulate into acc ----
    {
      const __bf16* Db = Dsb[blk];
      const int m = mwave * 16 + l15;
      for (int kc = 0; kc < 2; ++kc) {
        const int kb = kc * 32 + koff;
        v16bf af;
        #pragma unroll
        for (int i = 0; i < 8; ++i)
          af[i]     = (__bf16)(lds_u[m][kb + i]      * lds_v[m][kb + i]);
        #pragma unroll
        for (int i = 0; i < 8; ++i)
          af[8 + i] = (__bf16)(lds_u[m][kb + 16 + i] * lds_v[m][kb + 16 + i]);
        const __bf16* bp = Db + (((size_t)kc * 8 + nwave * 4) * 32 + lane) * 16;
        #pragma unroll
        for (int j = 0; j < 4; ++j)
          acc[j] = wmma_bf16(af, *(const v16bf*)(bp + (size_t)j * 512), acc[j]);
      }
      #pragma unroll
      for (int j = 0; j < 4; ++j)
        #pragma unroll
        for (int e = 0; e < 8; ++e)
          lds_r[crow + e][ccolb + j * 16 + l15] = acc[j][e];
    }
    __syncthreads();
  }

  // ---- unembed: out = r2 @ Wu^T, N=10 (tiny -> VALU) ----------------------
  for (int idx = tid; idx < MT * NOUT; idx += 256) {
    const int row = idx / NOUT;
    const int o   = idx % NOUT;
    const float* rp = lds_r[row];
    const float* wp = Wu + (size_t)o * DRES;
    float s = 0.f;
    #pragma unroll 8
    for (int k = 0; k < DRES; ++k) s += rp[k] * wp[k];
    out[(size_t)(rowBase + row) * NOUT + o] = s;
  }
}

// ---------------------------------------------------------------------------
extern "C" void kernel_launch(void* const* d_in, const int* in_sizes, int n_in,
                              void* d_out, int out_size, void* d_ws, size_t ws_size,
                              hipStream_t stream) {
  const float* x  = (const float*)d_in[0];
  const float* We = (const float*)d_in[1];
  const float* g1 = (const float*)d_in[2];
  const float* L1 = (const float*)d_in[3];
  const float* R1 = (const float*)d_in[4];
  const float* D1 = (const float*)d_in[5];
  const float* g2 = (const float*)d_in[6];
  const float* L2 = (const float*)d_in[7];
  const float* R2 = (const float*)d_in[8];
  const float* D2 = (const float*)d_in[9];
  const float* Wu = (const float*)d_in[10];
  float* out = (float*)d_out;

  const int rows = in_sizes[0] / KIN;   // 65536

  // workspace layout (bf16 fragment-packed weights), total ~864 KB
  char* ws = (char*)d_ws;
  size_t off = 0;
  __bf16* Wb  = (__bf16*)(ws + off); off += (size_t)KIN * DRES * sizeof(__bf16);
  __bf16* L1b = (__bf16*)(ws + off); off += (size_t)DRES * RANK * sizeof(__bf16);
  __bf16* R1b = (__bf16*)(ws + off); off += (size_t)DRES * RANK * sizeof(__bf16);
  __bf16* L2b = (__bf16*)(ws + off); off += (size_t)DRES * RANK * sizeof(__bf16);
  __bf16* R2b = (__bf16*)(ws + off); off += (size_t)DRES * RANK * sizeof(__bf16);
  __bf16* D1b = (__bf16*)(ws + off); off += (size_t)RANK * DRES * sizeof(__bf16);
  __bf16* D2b = (__bf16*)(ws + off); off += (size_t)RANK * DRES * sizeof(__bf16);
  (void)ws_size; (void)n_in; (void)out_size;

  auto pack = [&](const float* W, __bf16* dst, int Ktot, int Ntot) {
    int total = Ktot * Ntot;
    pack_b_frags<<<(total + 255) / 256, 256, 0, stream>>>(W, dst, Ktot, Ntot);
  };
  pack(We, Wb,  KIN,  DRES);   // B = We^T  [3072 x 128]
  pack(L1, L1b, DRES, RANK);   // B = L1^T  [128 x 64]
  pack(R1, R1b, DRES, RANK);
  pack(L2, L2b, DRES, RANK);
  pack(R2, R2b, DRES, RANK);
  pack(D1, D1b, RANK, DRES);   // B = D1^T  [64 x 128]
  pack(D2, D2b, RANK, DRES);

  fused_mlp<<<rows / MT, 256, 0, stream>>>(x, Wb, g1, L1b, R1b, D1b,
                                           g2, L2b, R2b, D2b, Wu, out);
}
